// MemoryAttention_87119116632751
// MI455X (gfx1250) — matrix operand order repaired and checked by reading, not compiled, not run on hardware
//
#include <hip/hip_runtime.h>

// ---------------------------------------------------------------------------
// MemoryAttention on gfx1250 (MI455X): f16 WMMA (f32 accum) for all GEMMs,
// single-pass softmax with f16 exp-cache in LDS, async global->LDS staging.
// Shapes fixed per reference: T=S=2048, B=2, E=QD=KD=1024, H=16, Dh=VD=64.
// ---------------------------------------------------------------------------

typedef _Float16 h16;
typedef __attribute__((ext_vector_type(16))) _Float16 v16h;
typedef __attribute__((ext_vector_type(8)))  _Float16 v8h;
typedef __attribute__((ext_vector_type(8)))  float    v8f;

#define T_DIM 2048
#define S_DIM 2048
#define B_DIM 2
#define E_DIM 1024
#define H_DIM 16
#define DH    64
#define VDIM  64
#define SCH   512   // softmax S-chunk staged in LDS (16 x 512 f32 = 32 KB)
#define NCH   (S_DIM / SCH)

#define WMMA16(a, b, c) \
  __builtin_amdgcn_wmma_f32_16x16x32_f16(false, (a), false, (b), (short)0, (c), false, false)

// ---------------------------------------------------------------------------
// CDNA5 async global->LDS copy (ASYNCcnt-tracked; see CDNA5 ISA).
// VDST = LDS byte offset VGPR, VADDR = 64-bit global address, no SADDR.
__device__ __forceinline__ void async_g2l_b128(void* lds_ptr, const void* gptr) {
  unsigned loff = (unsigned)(unsigned long long)lds_ptr;  // addr[31:0] = LDS offset
  asm volatile("global_load_async_to_lds_b128 %0, %1, off"
               :: "v"(loff), "v"(gptr) : "memory");
}
__device__ __forceinline__ void wait_async0() {
  asm volatile("s_wait_asynccnt 0x0" ::: "memory");
}

// ---- A fragment (16x32 MxK, f16 row-major source), CDNA5 wave32 layout:
// lanes 0-15: row=lane, K={k0..k0+7, k0+16..k0+23}; lanes 16-31: K shifted by 8.
__device__ __forceinline__ v16h load_a_frag_h(const h16* __restrict__ base, int ldm, int lane) {
  const int row  = lane & 15;
  const int ksel = (lane >> 4) << 3;          // 0 or 8
  const h16* p = base + row * ldm + ksel;
  v8h lo = *(const v8h*)(p);
  v8h hi = *(const v8h*)(p + 16);
  v16h r;
#pragma unroll
  for (int i = 0; i < 8; ++i) { r[i] = lo[i]; r[i + 8] = hi[i]; }
  return r;
}

// Same fragment, f16 source scaled by a per-row factor (softmax renorm).
__device__ __forceinline__ v16h load_a_frag_scaled(const h16* base, int ldm, int lane, float f) {
  const int row  = lane & 15;
  const int ksel = (lane >> 4) << 3;
  const h16* p = base + row * ldm + ksel;
  v8h lo = *(const v8h*)(p);
  v8h hi = *(const v8h*)(p + 16);
  const h16 hf = (h16)f;
  v16h r;
#pragma unroll
  for (int i = 0; i < 8; ++i) { r[i] = lo[i] * hf; r[i + 8] = hi[i] * hf; }
  return r;
}

// Same fragment but source is f32 (global query/key); convert to f16.
__device__ __forceinline__ v16h load_a_frag_f32(const float* __restrict__ base, int ldm, int lane) {
  const int row  = lane & 15;
  const int ksel = (lane >> 4) << 3;
  const float* p = base + row * ldm + ksel;
  v16h r;
#pragma unroll
  for (int i = 0; i < 8; ++i) { r[i] = (h16)p[i]; r[i + 8] = (h16)p[i + 16]; }
  return r;
}

// ---- B fragment (32x16 KxN). Source is row-major B^T (each column of B
// contiguous in K). lanes 0-15: col=lane, K={0..7, 16..23}; lanes 16-31: K shifted by 8.
__device__ __forceinline__ v16h load_b_frag_h(const h16* __restrict__ base, int ldn, int lane) {
  const int col  = lane & 15;
  const int ksel = (lane >> 4) << 3;          // 0 or 8
  const h16* p = base + col * ldn + ksel;
  v8h lo = *(const v8h*)(p);
  v8h hi = *(const v8h*)(p + 16);
  v16h r;
#pragma unroll
  for (int i = 0; i < 8; ++i) { r[i] = lo[i]; r[i + 8] = hi[i]; }
  return r;
}

// C/D 16x16 f32 layout: VGPR i -> row = i + 8*(lane>=16), col = lane&15.
__device__ __forceinline__ int c_row(int i, int lane) { return i + ((lane >> 4) << 3); }

// ---------------------------------------------------------------------------
__global__ __launch_bounds__(256) void k_cvt(const float* __restrict__ in,
                                             h16* __restrict__ out, int n) {
  int i = blockIdx.x * 256 + threadIdx.x;
  if (i < n) out[i] = (h16)in[i];
}

// value (S,B,VD) fp32 -> vt [b][vd][s] f16 (so P*V B-columns are contiguous)
__global__ __launch_bounds__(256) void k_vt(const float* __restrict__ val,
                                            h16* __restrict__ out) {
  int i = blockIdx.x * 256 + threadIdx.x;          // S*B*VD = 262144
  if (i >= S_DIM * B_DIM * VDIM) return;
  int s = i >> 7;
  int r = i & 127;
  int b = r >> 6, vd = r & 63;
  out[((b << 6) + vd) * S_DIM + s] = (h16)val[i];
}

// ---------------------------------------------------------------------------
// Q/K projection: out[(b*H+h)][t][dh] = ((X @ W^T)[m][e] + bias[e]) * scale
__global__ __launch_bounds__(256) void k_proj(const float* __restrict__ X,
                                              const h16* __restrict__ W,
                                              const float* __restrict__ bias,
                                              h16* __restrict__ out, float scale) {
  const int lane = threadIdx.x & 31;
  const int wave = threadIdx.x >> 5;
  const int m0 = blockIdx.x * 16;
  const int n0 = (blockIdx.y * 8 + wave) * 16;

  const float* Abase = X + (size_t)m0 * E_DIM + (lane & 15) * E_DIM + (((lane >> 4) << 3));
  const h16*   Bbase = W + (size_t)n0 * E_DIM + (lane & 15) * E_DIM + (((lane >> 4) << 4));

  v8f acc = {};
#pragma unroll 4
  for (int kk = 0; kk < E_DIM; kk += 32) {
    __builtin_prefetch(Abase + kk + 96, 0, 1);     // speculative: OOB is dropped
    __builtin_prefetch(Bbase + kk + 96, 0, 1);
    v16h a = load_a_frag_f32(X + (size_t)m0 * E_DIM + kk, E_DIM, lane);
    v16h b = load_b_frag_h(W + (size_t)n0 * E_DIM + kk, E_DIM, lane);
    acc = WMMA16(a, b, acc);
  }

  const int col  = lane & 15;
  const int e    = n0 + col;
  const int hsel = e >> 6, dh = e & 63;
  const float bv = bias[e];
#pragma unroll
  for (int i = 0; i < 8; ++i) {
    int m = m0 + c_row(i, lane);
    int t = m >> 1, b = m & 1;
    float v = (acc[i] + bv) * scale;
    out[(((b * H_DIM + hsel) * T_DIM + t) << 6) + dh] = (h16)v;
  }
}

// ---------------------------------------------------------------------------
// Output projection: d_out[m][e] = (ctx @ Wo^T)[m][e] + bo[e], fp32 out.
__global__ __launch_bounds__(256) void k_outproj(const h16* __restrict__ ctx,
                                                 const h16* __restrict__ Wo,
                                                 const float* __restrict__ bias,
                                                 float* __restrict__ out) {
  const int lane = threadIdx.x & 31;
  const int wave = threadIdx.x >> 5;
  const int m0 = blockIdx.x * 16;
  const int n0 = (blockIdx.y * 8 + wave) * 16;

  const h16* Abase = ctx + (size_t)m0 * E_DIM + (lane & 15) * E_DIM + (((lane >> 4) << 3));
  const h16* Bbase = Wo  + (size_t)n0 * E_DIM + (lane & 15) * E_DIM + (((lane >> 4) << 4));

  v8f acc = {};
#pragma unroll 4
  for (int kk = 0; kk < E_DIM; kk += 32) {
    __builtin_prefetch(Abase + kk + 96, 0, 1);
    __builtin_prefetch(Bbase + kk + 96, 0, 1);
    v16h a = load_a_frag_h(ctx + (size_t)m0 * E_DIM + kk, E_DIM, lane);
    v16h b = load_b_frag_h(Wo + (size_t)n0 * E_DIM + kk, E_DIM, lane);
    acc = WMMA16(a, b, acc);
  }

  const int col = lane & 15;
  const float bv = bias[n0 + col];
#pragma unroll
  for (int i = 0; i < 8; ++i) {
    int m = m0 + c_row(i, lane);
    out[(size_t)m * E_DIM + n0 + col] = acc[i] + bv;
  }
}

// ---------------------------------------------------------------------------
// Attention core. Grid: (T/16, B); block 256 = 8 wave32. One block owns a
// 16-row t-tile for one batch, loops all 16 heads (exclusive avg slice).
// Single pass: per chunk store exp(l - chunk_max) as f16 (64 KB LDS), then
// renormalize with per-chunk factors exp(cm_c - m)/sum when consuming.
__device__ __forceinline__ void logits_chunk(const h16* q, const h16* __restrict__ k,
                                             int sbase, float* sm_l,
                                             int wave, int lane) {
  for (int tile = wave; tile < SCH / 16; tile += 8) {
    v8f acc = {};
#pragma unroll
    for (int kk = 0; kk < DH; kk += 32) {
      v16h a = load_a_frag_h(q + kk, DH, lane);
      v16h b = load_b_frag_h(k + (size_t)(sbase + tile * 16) * DH + kk, DH, lane);
      acc = WMMA16(a, b, acc);
    }
    const int col = lane & 15;
#pragma unroll
    for (int i = 0; i < 8; ++i)
      sm_l[c_row(i, lane) * SCH + tile * 16 + col] = acc[i];
  }
}

__global__ __launch_bounds__(256) void k_attn(const h16* __restrict__ qh,
                                              const h16* __restrict__ kh,
                                              const h16* __restrict__ vt,
                                              h16* __restrict__ ctx,
                                              float* __restrict__ avg) {
  __shared__ __align__(16) h16   sm_e[16 * S_DIM];   // 64 KB exp cache (f16)
  __shared__ __align__(16) float sm_l[16 * SCH];     // 32 KB logits chunk
  __shared__ __align__(16) h16   sm_q[16 * DH];      // 2 KB q tile
  __shared__ float sm_red[256];
  __shared__ float sm_cm[NCH][16], sm_cs[NCH][16], sm_f[NCH][16];

  const int tid  = threadIdx.x;
  const int lane = tid & 31;
  const int wave = tid >> 5;
  const int t0 = blockIdx.x * 16;
  const int b  = blockIdx.y;
  const int row = tid >> 4, sub = tid & 15;          // 16 threads per t-row

  float* avg_base = avg + ((size_t)b * T_DIM + t0) * S_DIM;
  const float invH = 1.0f / (float)H_DIM;
  const h16* v = vt + (size_t)b * VDIM * S_DIM;

  for (int h = 0; h < H_DIM; ++h) {
    const int n = b * H_DIM + h;
    const h16* q_g = qh + ((size_t)n * T_DIM + t0) * DH;
    const h16* k   = kh + (size_t)n * S_DIM * DH;

    // stage this head's 16x64 q tile via async global->LDS (2 KB, 128 x b128)
    if (tid < 128) async_g2l_b128(&sm_q[tid * 8], q_g + tid * 8);
    wait_async0();
    __syncthreads();

    // ---------- logits -> exp(l - chunk_max) f16, plus chunk max/sum ----------
    for (int c = 0; c < NCH; ++c) {
      const int sbase = c * SCH;
      logits_chunk(sm_q, k, sbase, sm_l, wave, lane);
      __syncthreads();

      float lm = -1e30f;
      for (int j = sub; j < SCH; j += 16) lm = fmaxf(lm, sm_l[row * SCH + j]);
      sm_red[tid] = lm;
      __syncthreads();
      if (sub == 0) {
        float cm = sm_red[tid];
#pragma unroll
        for (int j = 1; j < 16; ++j) cm = fmaxf(cm, sm_red[tid + j]);
        sm_cm[c][row] = cm;
      }
      __syncthreads();

      const float cm = sm_cm[c][row];
      float ls = 0.0f;
      for (int j = sub; j < SCH; j += 16) {
        float e = __expf(sm_l[row * SCH + j] - cm);
        sm_e[row * S_DIM + sbase + j] = (h16)e;
        ls += e;
      }
      sm_red[tid] = ls;
      __syncthreads();
      if (sub == 0) {
        float cs = 0.0f;
#pragma unroll
        for (int j = 0; j < 16; ++j) cs += sm_red[tid + j];
        sm_cs[c][row] = cs;
      }
      __syncthreads();
    }

    // ---------- per-row renorm factors: f_c = exp(cm_c - m) / sum ----------
    if (tid < 16) {
      float m = sm_cm[0][tid];
#pragma unroll
      for (int c = 1; c < NCH; ++c) m = fmaxf(m, sm_cm[c][tid]);
      float s = 0.0f;
#pragma unroll
      for (int c = 0; c < NCH; ++c) s += sm_cs[c][tid] * __expf(sm_cm[c][tid] - m);
      const float inv = 1.0f / s;
#pragma unroll
      for (int c = 0; c < NCH; ++c) sm_f[c][tid] = __expf(sm_cm[c][tid] - m) * inv;
    }
    __syncthreads();

    // ---------- head-mean probs into d_out (block-exclusive slice) ----------
    for (int j = sub; j < S_DIM; j += 16) {
      float p = (float)sm_e[row * S_DIM + j] * sm_f[j >> 9][row];
      float* ap = avg_base + (size_t)row * S_DIM + j;
      float prev = (h == 0) ? 0.0f : *ap;
      *ap = prev + p * invH;
    }

    // ---------- P @ V over full S (8 waves = 4 VD-tiles x 2-way K split) ----
    v8f accpv = {};
    const int ntile = wave & 3;
    const int kpart = wave >> 2;
#pragma unroll 4
    for (int ks = kpart * (S_DIM / 2); ks < kpart * (S_DIM / 2) + S_DIM / 2; ks += 32) {
      const float f = sm_f[ks >> 9][lane & 15];                 // per-row factor
      v16h a  = load_a_frag_scaled(sm_e + ks, S_DIM, lane, f);  // probs (f16 LDS)
      v16h bb = load_b_frag_h(v + (size_t)(ntile * 16) * S_DIM + ks, S_DIM, lane);
      accpv = WMMA16(a, bb, accpv);
    }
    __syncthreads();   // sm_l free; reuse for partial combine

    {
      float* comb = sm_l;                 // 4 tiles * 256 floats
      if (kpart == 1) {
#pragma unroll
        for (int i = 0; i < 8; ++i) comb[ntile * 256 + lane * 8 + i] = accpv[i];
      }
      __syncthreads();
      if (kpart == 0) {
        const int col = lane & 15;
#pragma unroll
        for (int i = 0; i < 8; ++i) {
          float vsum = accpv[i] + comb[ntile * 256 + lane * 8 + i];
          int t  = t0 + c_row(i, lane);
          int vd = ntile * 16 + col;
          ctx[((size_t)(t * B_DIM + b) << 10) + h * 64 + vd] = (h16)vsum;
        }
      }
      __syncthreads();
    }
  }
}

// ---------------------------------------------------------------------------
extern "C" void kernel_launch(void* const* d_in, const int* in_sizes, int n_in,
                              void* d_out, int out_size, void* d_ws, size_t ws_size,
                              hipStream_t stream) {
  const float* query = (const float*)d_in[0];
  const float* key   = (const float*)d_in[1];
  const float* value = (const float*)d_in[2];
  const float* Wq    = (const float*)d_in[3];
  const float* bq    = (const float*)d_in[4];
  const float* Wk    = (const float*)d_in[5];
  const float* bk    = (const float*)d_in[6];
  const float* Wo    = (const float*)d_in[7];
  const float* bo    = (const float*)d_in[8];

  float* out_attn = (float*)d_out;                             // (T,B,1024)
  float* out_avg  = out_attn + (size_t)T_DIM * B_DIM * E_DIM;  // (B,T,S)

  h16* ws   = (h16*)d_ws;
  h16* wq_h = ws;                                        // 1M
  h16* wk_h = wq_h + (size_t)E_DIM * E_DIM;              // 1M
  h16* wo_h = wk_h + (size_t)E_DIM * E_DIM;              // 1M
  h16* qh   = wo_h + (size_t)E_DIM * E_DIM;              // 4M  [B*H][T][Dh]
  h16* kh   = qh + (size_t)B_DIM * H_DIM * T_DIM * DH;   // 4M
  h16* ctx  = kh + (size_t)B_DIM * H_DIM * S_DIM * DH;   // 4M  [t*B+b][E]
  h16* vt   = ctx + (size_t)T_DIM * B_DIM * E_DIM;       // 256K [b][vd][s]

  const int NW = E_DIM * E_DIM;
  k_cvt<<<dim3((NW + 255) / 256), 256, 0, stream>>>(Wq, wq_h, NW);
  k_cvt<<<dim3((NW + 255) / 256), 256, 0, stream>>>(Wk, wk_h, NW);
  k_cvt<<<dim3((NW + 255) / 256), 256, 0, stream>>>(Wo, wo_h, NW);

  const int NV = S_DIM * B_DIM * VDIM;
  k_vt<<<dim3((NV + 255) / 256), 256, 0, stream>>>(value, vt);

  const float scaling = 0.125f;                          // Dh^-0.5, Dh=64
  dim3 gproj(T_DIM * B_DIM / 16, E_DIM / (8 * 16));      // (256, 8)
  k_proj<<<gproj, 256, 0, stream>>>(query, wq_h, bq, qh, scaling);
  k_proj<<<gproj, 256, 0, stream>>>(key,   wk_h, bk, kh, 1.0f);

  dim3 gattn(T_DIM / 16, B_DIM);                         // (128, 2)
  k_attn<<<gattn, 256, 0, stream>>>(qh, kh, vt, ctx, out_avg);

  k_outproj<<<gproj, 256, 0, stream>>>(ctx, wo_h, bo, out_attn);
}
